// GAT_78116865180110
// MI455X (gfx1250) — matrix operand order repaired and checked by reading, not compiled, not run on hardware
//
#include <hip/hip_runtime.h>
#include <hip/hip_bf16.h>

#define NFEAT  50
#define NHID   64
#define NHEAD  4
#define NHEADO 6
#define NCLASS 121
#define ALPHA  0.2f

typedef __attribute__((ext_vector_type(16))) _Float16 v16h;
typedef __attribute__((ext_vector_type(8)))  _Float16 h8;
typedef __attribute__((ext_vector_type(8)))  float    v8f;

// ---------------------------------------------------------------------------
// Conversion / padding kernels (run once per layer / per head, bandwidth-bound)
// ---------------------------------------------------------------------------
// Xh[i, k] = (k < fin) ? X[i*fin + k] : 0   with row stride fin_pad, f16
__global__ void k_cvt_x(const float* __restrict__ X, _Float16* __restrict__ Xh,
                        int n, int fin, int fin_pad)
{
    const long long idx = (long long)blockIdx.x * blockDim.x + threadIdx.x;
    if (idx >= (long long)n * fin_pad) return;
    const int i = (int)(idx / fin_pad);
    const int k = (int)(idx % fin_pad);
    const float v = (k < fin) ? X[(long long)i * fin + k] : 0.f;
    Xh[idx] = (_Float16)v;
}

// WTh[c, k] = (c < fout && k < fin) ? W[k*fout + c] : 0   (transposed, padded, f16)
__global__ void k_cvt_wT(const float* __restrict__ W, _Float16* __restrict__ WTh,
                         int fin, int fout, int fin_pad, int fout_pad)
{
    const int idx = blockIdx.x * blockDim.x + threadIdx.x;
    if (idx >= fout_pad * fin_pad) return;
    const int c = idx / fin_pad;
    const int k = idx % fin_pad;
    const float v = (c < fout && k < fin) ? W[(long long)k * fout + c] : 0.f;
    WTh[idx] = (_Float16)v;
}

// ---------------------------------------------------------------------------
// GEMM: H[n x FOUT] = Xh[n x FIN_PAD] @ WTh^T, f16 WMMA, f32 accumulate.
// One wave computes a 16-row stripe x (NT*16) columns; A-frag reused NT times.
// Requires n % 16 == 0 (true here: 50000 = 3125*16).
// A-frag (16x32 f16): lane l: M=l&15, Kb=(l<16?0:8); halves Kb..Kb+7, Kb+16..Kb+23.
// B-frag (32x16 f16): lane l: Nc=l&15, Kb=(l<16?0:8); halves Kb..Kb+7, Kb+16..Kb+23.
// C/D: VGPR r -> M=r+(l<16?0:8), N=l&15.
// ---------------------------------------------------------------------------
template<int FIN_PAD, int NT, int FOUT>
__global__ void k_gemm_wmma(const _Float16* __restrict__ Xh, const _Float16* __restrict__ WTh,
                            float* __restrict__ H, int n)
{
    const int lane  = threadIdx.x & 31;
    const int wave  = (blockIdx.x * blockDim.x + threadIdx.x) >> 5;
    const int waves = (gridDim.x * blockDim.x) >> 5;
    const int rowTiles = n >> 4;

    const int mA    = lane & 15;
    const int kOffA = (lane < 16) ? 0 : 8;
    const int nB    = lane & 15;
    const int kOffB = (lane < 16) ? 0 : 8;
    const int mOffC = (lane < 16) ? 0 : 8;

    for (int tm = wave; tm < rowTiles; tm += waves) {
        const int row0 = tm << 4;
        const _Float16* aPtr = Xh + (size_t)(row0 + mA) * FIN_PAD + kOffA;

        v8f acc[NT];
#pragma unroll
        for (int t = 0; t < NT; ++t) acc[t] = (v8f){};

#pragma unroll
        for (int k0 = 0; k0 < FIN_PAD; k0 += 32) {
            const h8 alo = *(const h8*)(aPtr + k0);
            const h8 ahi = *(const h8*)(aPtr + k0 + 16);
            const v16h af = __builtin_shufflevector(alo, ahi,
                0, 1, 2, 3, 4, 5, 6, 7, 8, 9, 10, 11, 12, 13, 14, 15);
#pragma unroll
            for (int t = 0; t < NT; ++t) {
                const _Float16* bPtr = WTh + (size_t)(t * 16 + nB) * FIN_PAD + kOffB + k0;
                const h8 blo = *(const h8*)(bPtr);
                const h8 bhi = *(const h8*)(bPtr + 16);
                const v16h bf = __builtin_shufflevector(blo, bhi,
                    0, 1, 2, 3, 4, 5, 6, 7, 8, 9, 10, 11, 12, 13, 14, 15);
                acc[t] = __builtin_amdgcn_wmma_f32_16x16x32_f16(
                    false, af, false, bf, (short)0, acc[t], false, false);
            }
        }

        // store: base pointer once per column tile; row offsets are immediates
#pragma unroll
        for (int t = 0; t < NT; ++t) {
            const int col = t * 16 + nB;
            if (NT * 16 == FOUT || col < FOUT) {
                float* cPtr = H + (size_t)(row0 + mOffC) * FOUT + col;
#pragma unroll
                for (int r = 0; r < 8; ++r)
                    cPtr[(size_t)r * FOUT] = acc[t][r];
            }
        }
    }
}

// e_s[i] = h[i] . a[0:F],  e_t[i] = h[i] . a[F:2F]
__global__ void k_node_scores(const float* __restrict__ H, const float* __restrict__ a,
                              float* __restrict__ es, float* __restrict__ et, int n, int F)
{
    const int i = blockIdx.x * blockDim.x + threadIdx.x;
    if (i >= n) return;
    const float* hr = H + (long long)i * F;
    float s = 0.f, t = 0.f;
    for (int f = 0; f < F; ++f) {
        const float hv = hr[f];
        s += hv * a[f];
        t += hv * a[F + f];
    }
    es[i] = s;
    et[i] = t;
}

__device__ __forceinline__ int f2key(float f) {
    const int b = __float_as_int(f);
    return b ^ ((b >> 31) & 0x7fffffff);
}
__device__ __forceinline__ float key2f(int k) {
    const int b = k ^ ((k >> 31) & 0x7fffffff);
    return __int_as_float(b);
}

__global__ void k_edge_logits(const int* __restrict__ src, const int* __restrict__ tgt,
                              const float* __restrict__ es, const float* __restrict__ et,
                              float* __restrict__ elog, int* __restrict__ gmax, int E)
{
    const int e = blockIdx.x * blockDim.x + threadIdx.x;
    if (e >= E) return;
    float v = es[src[e]] + et[tgt[e]];
    v = (v > 0.f) ? v : ALPHA * v;
    elog[e] = v;
    atomicMax(gmax, f2key(v));
}

__global__ void k_edge_exp(const int* __restrict__ src, float* __restrict__ ew,
                           const int* __restrict__ gmax, float* __restrict__ ssum, int E)
{
    const int e = blockIdx.x * blockDim.x + threadIdx.x;
    if (e >= E) return;
    const float m = key2f(*gmax);
    const float w = __expf(ew[e] - m);
    ew[e] = w;
    atomicAdd(&ssum[src[e]], w);
}

// wave per edge: AGG[src] += (ew[e]/ssum[src]) * H[tgt]
__global__ void k_edge_agg(const int* __restrict__ src, const int* __restrict__ tgt,
                           const float* __restrict__ ew, const float* __restrict__ ssum,
                           const float* __restrict__ H, float* __restrict__ AGG, int E, int F)
{
    const int lane = threadIdx.x & 31;
    const int wid  = (blockIdx.x * blockDim.x + threadIdx.x) >> 5;
    const int nw   = (gridDim.x * blockDim.x) >> 5;
    for (int e = wid; e < E; e += nw) {
        const int s = src[e], d = tgt[e];
        const float att = ew[e] / ssum[s];
        const float* hd = H + (long long)d * F;
        float* os = AGG + (long long)s * F;
        for (int f = lane; f < F; f += 32)
            atomicAdd(os + f, att * hd[f]);
    }
}

// xout[i, colOff + f] = elu(agg[i,f] + b[f])
__global__ void k_elu_store(const float* __restrict__ agg, const float* __restrict__ b,
                            float* __restrict__ xout, int n, int F, int stride, int colOff)
{
    const int idx = blockIdx.x * blockDim.x + threadIdx.x;
    if (idx >= n * F) return;
    const int i = idx / F, f = idx % F;
    float v = agg[(long long)i * F + f] + b[f];
    v = (v > 0.f) ? v : (__expf(v) - 1.f);
    xout[(long long)i * stride + colOff + f] = v;
}

// out[i,f] += (agg[i,f] + b[f]) * scale
__global__ void k_accum(const float* __restrict__ agg, const float* __restrict__ b,
                        float* __restrict__ out, int n, int F, float scale)
{
    const int idx = blockIdx.x * blockDim.x + threadIdx.x;
    if (idx >= n * F) return;
    const int i = idx / F, f = idx % F;
    out[idx] += (agg[(long long)i * F + f] + b[f]) * scale;
}

__global__ void k_sigmoid(float* __restrict__ p, int n)
{
    const int i = blockIdx.x * blockDim.x + threadIdx.x;
    if (i >= n) return;
    p[i] = 1.f / (1.f + __expf(-p[i]));
}

__global__ void k_zero(float* __restrict__ p, long long n)
{
    const long long i = (long long)blockIdx.x * blockDim.x + threadIdx.x;
    if (i < n) p[i] = 0.f;
}

__global__ void k_set_int(int* p, int v) { *p = v; }

// ---------------------------------------------------------------------------
static inline int cdiv(long long a, long long b) { return (int)((a + b - 1) / b); }

extern "C" void kernel_launch(void* const* d_in, const int* in_sizes, int n_in,
                              void* d_out, int out_size, void* d_ws, size_t ws_size,
                              hipStream_t stream)
{
    const float* x   = (const float*)d_in[0];
    const int*   el  = (const int*)  d_in[1];
    const float* W1  = (const float*)d_in[2];
    const float* a1  = (const float*)d_in[3];
    const float* b1  = (const float*)d_in[4];
    const float* W2  = (const float*)d_in[5];
    const float* a2  = (const float*)d_in[6];
    const float* b2  = (const float*)d_in[7];
    const float* W3  = (const float*)d_in[8];
    const float* a3  = (const float*)d_in[9];
    const float* b3  = (const float*)d_in[10];

    const int N = in_sizes[0] / NFEAT;
    const int E = in_sizes[1] / 2;
    const int* src = el;
    const int* tgt = el + E;

    // workspace layout: f16 buffers first (16B-aligned), then f32, then scalar
    _Float16* Xh  = (_Float16*)d_ws;                  // N*256 halves
    _Float16* WTh = Xh + (long long)N * 256;          // 128*256 halves
    float* x1   = (float*)(WTh + 128 * 256);          // N*256
    float* x2   = x1 + (long long)N * 256;            // N*256
    float* h    = x2 + (long long)N * 256;            // N*128
    float* agg  = h  + (long long)N * 128;            // N*128
    float* es   = agg + (long long)N * 128;           // N
    float* et   = es + N;                             // N
    float* ssum = et + N;                             // N
    float* ew   = ssum + N;                           // E
    int*   gmax = (int*)(ew + E);                     // 1
    (void)ws_size; (void)n_in; (void)out_size;

    const int T = 256;
    const int aggBlocks  = cdiv((long long)E * 32, T);
    const int gemmBlocks = cdiv(((long long)N / 16) * 32, T);

    // attention tail shared by every head (h already computed into `h`)
    auto run_attention = [&](const float* a, int fout) {
        k_node_scores<<<cdiv(N, T), T, 0, stream>>>(h, a, es, et, N, fout);
        k_zero<<<cdiv(N, T), T, 0, stream>>>(ssum, N);
        k_zero<<<cdiv((long long)N * fout, T), T, 0, stream>>>(agg, (long long)N * fout);
        k_set_int<<<1, 1, 0, stream>>>(gmax, (int)0x80000000);
        k_edge_logits<<<cdiv(E, T), T, 0, stream>>>(src, tgt, es, et, ew, gmax, E);
        k_edge_exp<<<cdiv(E, T), T, 0, stream>>>(src, ew, gmax, ssum, E);
        k_edge_agg<<<aggBlocks, T, 0, stream>>>(src, tgt, ew, ssum, h, agg, E, fout);
    };

    // ---- layer 1: fin=50 (pad 64), fout=64, 4 heads -> x1 (N x 256), ELU ----
    k_cvt_x<<<cdiv((long long)N * 64, T), T, 0, stream>>>(x, Xh, N, NFEAT, 64);
    for (int hd = 0; hd < NHEAD; ++hd) {
        k_cvt_wT<<<cdiv(64 * 64, T), T, 0, stream>>>(
            W1 + (long long)hd * NFEAT * NHID, WTh, NFEAT, NHID, 64, 64);
        k_gemm_wmma<64, 4, 64><<<gemmBlocks, T, 0, stream>>>(Xh, WTh, h, N);
        run_attention(a1 + (long long)hd * 2 * NHID, NHID);
        k_elu_store<<<cdiv((long long)N * NHID, T), T, 0, stream>>>(
            agg, b1 + (long long)hd * NHID, x1, N, NHID, NHEAD * NHID, hd * NHID);
    }

    // ---- layer 2: fin=256, fout=64, 4 heads -> x2 (N x 256), ELU ----
    k_cvt_x<<<cdiv((long long)N * 256, T), T, 0, stream>>>(x1, Xh, N, 256, 256);
    for (int hd = 0; hd < NHEAD; ++hd) {
        k_cvt_wT<<<cdiv(64 * 256, T), T, 0, stream>>>(
            W2 + (long long)hd * 256 * NHID, WTh, 256, NHID, 256, 64);
        k_gemm_wmma<256, 4, 64><<<gemmBlocks, T, 0, stream>>>(Xh, WTh, h, N);
        run_attention(a2 + (long long)hd * 2 * NHID, NHID);
        k_elu_store<<<cdiv((long long)N * NHID, T), T, 0, stream>>>(
            agg, b2 + (long long)hd * NHID, x2, N, NHID, NHEAD * NHID, hd * NHID);
    }

    // ---- layer 3: fin=256, fout=121 (pad 128), 6 heads -> mean -> sigmoid ----
    float* out = (float*)d_out;
    k_zero<<<cdiv((long long)N * NCLASS, T), T, 0, stream>>>(out, (long long)N * NCLASS);
    k_cvt_x<<<cdiv((long long)N * 256, T), T, 0, stream>>>(x2, Xh, N, 256, 256);
    for (int hd = 0; hd < NHEADO; ++hd) {
        k_cvt_wT<<<cdiv(128 * 256, T), T, 0, stream>>>(
            W3 + (long long)hd * 256 * NCLASS, WTh, 256, NCLASS, 256, 128);
        k_gemm_wmma<256, 8, 121><<<gemmBlocks, T, 0, stream>>>(Xh, WTh, h, N);
        run_attention(a3 + (long long)hd * 2 * NCLASS, NCLASS);
        k_accum<<<cdiv((long long)N * NCLASS, T), T, 0, stream>>>(
            agg, b3 + (long long)hd * NCLASS, out, N, NCLASS, 1.0f / NHEADO);
    }
    k_sigmoid<<<cdiv((long long)N * NCLASS, T), T, 0, stream>>>(out, N * NCLASS);
}
